// UpFlameAGOMoE_75445395521785
// MI455X (gfx1250) — matrix operand order repaired and checked by reading, not compiled, not run on hardware
//
#include <hip/hip_runtime.h>
#include <stdint.h>

// ---------------- problem constants (fixed by reference) ----------------
constexpr int H_    = 1024;   // hidden
constexpr int I_    = 3584;   // intermediate
constexpr int E_    = 8;      // experts
constexpr int NTOK  = 8192;   // B*S = 4*2048

// ---------------- types ----------------
typedef __bf16 bf16_t;
typedef __bf16 v16bf __attribute__((ext_vector_type(16)));
typedef __bf16 v8bf  __attribute__((ext_vector_type(8)));
typedef float  v8f   __attribute__((ext_vector_type(8)));

union FragBF { v16bf v; v8bf h[2]; };

__device__ __forceinline__ unsigned short f2bf_bits(float f) {
  union { float fl; uint32_t u; } c; c.fl = f;
  uint32_t u = c.u;
  uint32_t r = u + 0x7FFFu + ((u >> 16) & 1u);   // round-to-nearest-even
  return (unsigned short)(r >> 16);
}
__device__ __forceinline__ bf16_t f2bf(float f) {
  union { unsigned short s; bf16_t b; } c; c.s = f2bf_bits(f);
  return c.b;
}

// ---- CDNA5 async global->LDS copy (ASYNCcnt path, ISA §10 / §15.18) ----
// INST_OFFSET is added to BOTH the LDS dest and the global address (ISA §4.4),
// so one immediate moves both pointers together.
template <int OFF>
__device__ __forceinline__ void async_b128_o(uint32_t lds_off, const void* gptr) {
  asm volatile("global_load_async_to_lds_b128 %0, %1, off offset:%2"
               :: "v"(lds_off), "v"((uint64_t)(uintptr_t)gptr), "n"(OFF)
               : "memory");
}
__device__ __forceinline__ void wait_async0() {
  asm volatile("s_wait_asynccnt 0" ::: "memory");
}
__device__ __forceinline__ uint32_t lds_addr32(const void* p) {
  return (uint32_t)(uintptr_t)p;   // LDS aperture: offset lives in addr[31:0]
}

// ---------------- fp32 -> bf16 bulk convert (x and per-expert weights) ----
__global__ __launch_bounds__(256)
void cvt_bf16_kernel(const float* __restrict__ src, bf16_t* __restrict__ dst) {
  size_t i = ((size_t)blockIdx.x * blockDim.x + threadIdx.x) * 4;
  float4 f = *(const float4*)(src + i);
  ushort4 u = make_ushort4(f2bf_bits(f.x), f2bf_bits(f.y), f2bf_bits(f.z), f2bf_bits(f.w));
  *(ushort4*)((unsigned short*)dst + i) = u;
}

// ---------------- router: logits, softmax, top-2, renorm ----------------
__global__ __launch_bounds__(256)
void router_kernel(const float* __restrict__ x, const float* __restrict__ gw,
                   float* __restrict__ logits_out, float* __restrict__ comb) {
  const int tok  = blockIdx.x * 8 + (threadIdx.x >> 5);
  const int lane = threadIdx.x & 31;
  float acc[E_];
  #pragma unroll
  for (int e = 0; e < E_; ++e) acc[e] = 0.0f;
  const float* xr = x + (size_t)tok * H_;
  for (int h = lane; h < H_; h += 32) {
    float xv = xr[h];
    #pragma unroll
    for (int e = 0; e < E_; ++e) acc[e] += xv * gw[e * H_ + h];
  }
  #pragma unroll
  for (int e = 0; e < E_; ++e)
    #pragma unroll
    for (int off = 16; off > 0; off >>= 1) acc[e] += __shfl_xor(acc[e], off, 32);

  if (lane == 0) {
    float mx = acc[0];
    #pragma unroll
    for (int e = 1; e < E_; ++e) mx = fmaxf(mx, acc[e]);
    float p[E_], sum = 0.0f;
    #pragma unroll
    for (int e = 0; e < E_; ++e) { p[e] = __expf(acc[e] - mx); sum += p[e]; }
    #pragma unroll
    for (int e = 0; e < E_; ++e) p[e] /= sum;
    int i1 = 0;
    #pragma unroll
    for (int e = 1; e < E_; ++e) if (p[e] > p[i1]) i1 = e;
    int i2 = (i1 == 0) ? 1 : 0;
    #pragma unroll
    for (int e = 0; e < E_; ++e) if (e != i1 && p[e] > p[i2]) i2 = e;
    float s = p[i1] + p[i2];
    float w1 = p[i1] / s, w2 = p[i2] / s;
    #pragma unroll
    for (int e = 0; e < E_; ++e) {
      logits_out[(size_t)tok * E_ + e] = acc[e];
      comb[(size_t)e * NTOK + tok] = (e == i1) ? w1 : ((e == i2) ? w2 : 0.0f);
    }
  }
}

// ---------------- GEMM1: h = silu(x @ Wg^T) * (x @ Wu^T), all bf16 in ----
// block tile 128(M) x 64(N), BK=32, double-buffered LDS; 4 waves (2x2), 64x32/wave
__global__ __launch_bounds__(128)
void moe_gateup(const bf16_t* __restrict__ xb,
                const bf16_t* __restrict__ Wg,
                const bf16_t* __restrict__ Wu,
                bf16_t* __restrict__ hbuf) {
  constexpr int PK = 40;  // padded LDS row (80B = 5*16B, keeps 16B alignment)
  __shared__ __align__(16) bf16_t As [2][128][PK];
  __shared__ __align__(16) bf16_t Bgs[2][64][PK];
  __shared__ __align__(16) bf16_t Bus[2][64][PK];

  const int tid  = threadIdx.x;
  const int lane = tid & 31;
  const int wid  = tid >> 5;
  const int wm   = (wid >> 1) * 64;
  const int wn   = (wid & 1) * 32;
  const int m0   = blockIdx.y * 128;
  const int n0   = blockIdx.x * 64;

  const int frow  = lane & 15;
  const int abase = (lane < 16) ? 0 : 8;   // A frag K chunks: {abase..+7, abase+16..+23}
  const int boff  = (lane < 16) ? 0 : 8;   // B frag K chunks: {boff..+7, boff+16..+23}

  // per-thread staging addresses (one LDS base per buffer; chunk offsets immediate)
  const bf16_t* aSrcRow = xb + (size_t)(m0 + tid) * H_;
  const bf16_t* bSrcRow = ((tid < 64) ? Wg : Wu) + (size_t)(n0 + (tid & 63)) * H_;
  const uint32_t aLds[2] = { lds_addr32(&As[0][tid][0]), lds_addr32(&As[1][tid][0]) };
  const uint32_t bLds[2] = {
      lds_addr32((tid < 64) ? &Bgs[0][tid & 63][0] : &Bus[0][tid & 63][0]),
      lds_addr32((tid < 64) ? &Bgs[1][tid & 63][0] : &Bus[1][tid & 63][0]) };

  auto issue = [&](int buf, int k0) {
    const bf16_t* aS = aSrcRow + k0;
    const bf16_t* bS = bSrcRow + k0;
    async_b128_o< 0>(aLds[buf], aS);
    async_b128_o<16>(aLds[buf], aS);
    async_b128_o<32>(aLds[buf], aS);
    async_b128_o<48>(aLds[buf], aS);
    async_b128_o< 0>(bLds[buf], bS);
    async_b128_o<16>(bLds[buf], bS);
    async_b128_o<32>(bLds[buf], bS);
    async_b128_o<48>(bLds[buf], bS);
  };

  v8f accG[4][2], accU[4][2];
  #pragma unroll
  for (int i = 0; i < 4; ++i)
    #pragma unroll
    for (int j = 0; j < 2; ++j) { accG[i][j] = 0.0f; accU[i][j] = 0.0f; }

  auto compute = [&](int buf) {
    #pragma unroll
    for (int mi = 0; mi < 4; ++mi) {
      FragBF a;
      const bf16_t* ar = &As[buf][wm + mi * 16 + frow][0];
      a.h[0] = *(const v8bf*)(ar + abase);
      a.h[1] = *(const v8bf*)(ar + abase + 16);
      #pragma unroll
      for (int ni = 0; ni < 2; ++ni) {
        FragBF bg, bu;
        const bf16_t* bgr = &Bgs[buf][wn + ni * 16 + frow][0];
        const bf16_t* bur = &Bus[buf][wn + ni * 16 + frow][0];
        bg.h[0] = *(const v8bf*)(bgr + boff);
        bg.h[1] = *(const v8bf*)(bgr + boff + 16);
        bu.h[0] = *(const v8bf*)(bur + boff);
        bu.h[1] = *(const v8bf*)(bur + boff + 16);
        accG[mi][ni] = __builtin_amdgcn_wmma_f32_16x16x32_bf16(
            false, a.v, false, bg.v, (short)0, accG[mi][ni], false, false);
        accU[mi][ni] = __builtin_amdgcn_wmma_f32_16x16x32_bf16(
            false, a.v, false, bu.v, (short)0, accU[mi][ni], false, false);
      }
    }
  };

  // ping-pong pipeline: one barrier per k-step; the barrier at the top of each
  // step proves every wave finished reading the buffer we are about to refill.
  issue(0, 0);
  for (int k0 = 0; k0 < H_; k0 += 64) {
    wait_async0();
    __syncthreads();
    issue(1, k0 + 32);
    compute(0);
    wait_async0();
    __syncthreads();
    if (k0 + 64 < H_) issue(0, k0 + 64);
    compute(1);
  }

  // epilogue: SiLU(g)*u -> bf16. D layout: col = lane&15, row = vgpr + 8*(lane>=16)
  const int mrow0 = m0 + wm + ((lane >= 16) ? 8 : 0);
  const int ncol0 = n0 + wn + (lane & 15);
  #pragma unroll
  for (int mi = 0; mi < 4; ++mi)
    #pragma unroll
    for (int ni = 0; ni < 2; ++ni)
      #pragma unroll
      for (int r = 0; r < 8; ++r) {
        float g = accG[mi][ni][r];
        float u = accU[mi][ni][r];
        float hv = g * __builtin_amdgcn_rcpf(1.0f + __expf(-g)) * u;
        hbuf[(size_t)(mrow0 + mi * 16 + r) * I_ + (ncol0 + ni * 16)] = f2bf(hv);
      }
}

// ---------------- GEMM2: out (+)= (h @ Wd^T) * comb[token] ----------------
template <bool FIRST>
__global__ __launch_bounds__(128)
void moe_down(const bf16_t* __restrict__ hbuf,
              const bf16_t* __restrict__ Wd,
              const float* __restrict__ combE,
              float* __restrict__ out) {
  constexpr int PK = 40;
  __shared__ __align__(16) bf16_t As[2][128][PK];
  __shared__ __align__(16) bf16_t Bs[2][64][PK];

  const int tid  = threadIdx.x;
  const int lane = tid & 31;
  const int wid  = tid >> 5;
  const int wm   = (wid >> 1) * 64;
  const int wn   = (wid & 1) * 32;
  const int m0   = blockIdx.y * 128;
  const int n0   = blockIdx.x * 64;

  const int frow  = lane & 15;
  const int abase = (lane < 16) ? 0 : 8;
  const int boff  = (lane < 16) ? 0 : 8;

  const bf16_t* aSrcRow = hbuf + (size_t)(m0 + tid) * I_;
  const int     br = tid >> 1, bh = tid & 1;             // 2 threads per B row
  const bf16_t* bSrcRow = Wd + (size_t)(n0 + br) * I_ + bh * 16;
  const uint32_t aLds[2] = { lds_addr32(&As[0][tid][0]), lds_addr32(&As[1][tid][0]) };
  const uint32_t bLds[2] = { lds_addr32(&Bs[0][br][bh * 16]), lds_addr32(&Bs[1][br][bh * 16]) };

  auto issue = [&](int buf, int k0) {
    const bf16_t* aS = aSrcRow + k0;
    const bf16_t* bS = bSrcRow + k0;
    async_b128_o< 0>(aLds[buf], aS);
    async_b128_o<16>(aLds[buf], aS);
    async_b128_o<32>(aLds[buf], aS);
    async_b128_o<48>(aLds[buf], aS);
    async_b128_o< 0>(bLds[buf], bS);
    async_b128_o<16>(bLds[buf], bS);
  };

  v8f acc[4][2];
  #pragma unroll
  for (int i = 0; i < 4; ++i)
    #pragma unroll
    for (int j = 0; j < 2; ++j) acc[i][j] = 0.0f;

  auto compute = [&](int buf) {
    #pragma unroll
    for (int mi = 0; mi < 4; ++mi) {
      FragBF a;
      const bf16_t* ar = &As[buf][wm + mi * 16 + frow][0];
      a.h[0] = *(const v8bf*)(ar + abase);
      a.h[1] = *(const v8bf*)(ar + abase + 16);
      #pragma unroll
      for (int ni = 0; ni < 2; ++ni) {
        FragBF b;
        const bf16_t* brp = &Bs[buf][wn + ni * 16 + frow][0];
        b.h[0] = *(const v8bf*)(brp + boff);
        b.h[1] = *(const v8bf*)(brp + boff + 16);
        acc[mi][ni] = __builtin_amdgcn_wmma_f32_16x16x32_bf16(
            false, a.v, false, b.v, (short)0, acc[mi][ni], false, false);
      }
    }
  };

  issue(0, 0);
  for (int k0 = 0; k0 < I_; k0 += 64) {
    wait_async0();
    __syncthreads();
    issue(1, k0 + 32);
    compute(0);
    wait_async0();
    __syncthreads();
    if (k0 + 64 < I_) issue(0, k0 + 64);
    compute(1);
  }

  const int mrow0 = m0 + wm + ((lane >= 16) ? 8 : 0);
  const int ncol0 = n0 + wn + (lane & 15);
  #pragma unroll
  for (int mi = 0; mi < 4; ++mi)
    #pragma unroll
    for (int r = 0; r < 8; ++r) {
      const int m = mrow0 + mi * 16 + r;
      const float cw = combE[m];
      #pragma unroll
      for (int ni = 0; ni < 2; ++ni) {
        size_t o = (size_t)m * H_ + (ncol0 + ni * 16);
        float v = acc[mi][ni][r] * cw;
        if (FIRST) out[o] = v; else out[o] += v;
      }
    }
}

// ---------------- host launch ----------------
extern "C" void kernel_launch(void* const* d_in, const int* in_sizes, int n_in,
                              void* d_out, int out_size, void* d_ws, size_t ws_size,
                              hipStream_t stream) {
  (void)in_sizes; (void)n_in; (void)out_size; (void)ws_size;
  const float* x  = (const float*)d_in[0];  // [NTOK, H]
  const float* gw = (const float*)d_in[1];  // [E, H]
  const float* Wg = (const float*)d_in[2];  // [E, I, H]
  const float* Wu = (const float*)d_in[3];  // [E, I, H]
  const float* Wd = (const float*)d_in[4];  // [E, H, I]
  // d_in[5] = top_k (hardcoded 2)

  float* out    = (float*)d_out;                 // [NTOK*H]
  float* logits = out + (size_t)NTOK * H_;       // [NTOK*E]

  char* ws = (char*)d_ws;
  const size_t XB = (size_t)NTOK * H_ * sizeof(bf16_t);   // 16.8 MB
  const size_t HB = (size_t)NTOK * I_ * sizeof(bf16_t);   // 58.7 MB
  const size_t CB = (size_t)E_ * NTOK * sizeof(float);    // 0.26 MB
  const size_t WB = (size_t)I_ * H_ * sizeof(bf16_t);     // 7.34 MB per matrix
  bf16_t* xb   = (bf16_t*)ws;
  bf16_t* hb   = (bf16_t*)(ws + XB);
  float*  comb = (float*)(ws + XB + HB);
  bf16_t* wgb  = (bf16_t*)(ws + XB + HB + CB);
  bf16_t* wub  = (bf16_t*)(ws + XB + HB + CB + WB);
  bf16_t* wdb  = (bf16_t*)(ws + XB + HB + CB + 2 * WB);

  const int cvtBlkX = (int)((size_t)NTOK * H_ / 1024);   // 4 elems/thread, 256 thr
  const int cvtBlkW = (int)((size_t)I_ * H_ / 1024);

  cvt_bf16_kernel<<<cvtBlkX, 256, 0, stream>>>(x, xb);
  router_kernel<<<NTOK / 8, 256, 0, stream>>>(x, gw, logits, comb);

  dim3 g1(I_ / 64, NTOK / 128);
  dim3 g2(H_ / 64, NTOK / 128);
  for (int e = 0; e < E_; ++e) {
    cvt_bf16_kernel<<<cvtBlkW, 256, 0, stream>>>(Wg + (size_t)e * I_ * H_, wgb);
    cvt_bf16_kernel<<<cvtBlkW, 256, 0, stream>>>(Wu + (size_t)e * I_ * H_, wub);
    moe_gateup<<<g1, 128, 0, stream>>>(xb, wgb, wub, hb);
    cvt_bf16_kernel<<<cvtBlkW, 256, 0, stream>>>(Wd + (size_t)e * H_ * I_, wdb);
    if (e == 0)
      moe_down<true ><<<g2, 128, 0, stream>>>(hb, wdb, comb + (size_t)e * NTOK, out);
    else
      moe_down<false><<<g2, 128, 0, stream>>>(hb, wdb, comb + (size_t)e * NTOK, out);
  }
}
